// SplineTransformer_5583457485594
// MI455X (gfx1250) — matrix operand order repaired and checked by reading, not compiled, not run on hardware
//
#include <hip/hip_runtime.h>
#include <hip/hip_bf16.h>
#include <stddef.h>
#include <stdint.h>

typedef __attribute__((ext_vector_type(16))) __bf16 v16bf;
typedef __attribute__((ext_vector_type(8)))  __bf16 v8bf;
typedef __attribute__((ext_vector_type(8)))  float  v8f;

#define B_SZ   16384
#define DX_SZ  128
#define DY_SZ  128
#define H_SZ   2048
#define K_SP   16
#define NCOEF  (3 * K_SP + 1)           // 49
#define NOUT   (DY_SZ * NCOEF)          // 6272
#define MIN_BIN   0.001f
#define MIN_DERIV 0.001f

// ---------------- f32 -> bf16 conversion (x only) ----------------
__global__ void spline_cvt_bf16(const float* __restrict__ src,
                                __bf16* __restrict__ dst, int n) {
    int i = blockIdx.x * blockDim.x + threadIdx.x;
    int stride = gridDim.x * blockDim.x;
    for (; i < n; i += stride) dst[i] = (__bf16)src[i];
}

// ---------------- f32 [K][N] -> bf16 [N][K] transpose+convert ----------------
__global__ __launch_bounds__(256)
void spline_cvtT_bf16(const float* __restrict__ src, __bf16* __restrict__ dst,
                      int Kd, int N) {
    __shared__ float tile[32][33];
    const int tx = threadIdx.x & 31;
    const int ty = threadIdx.x >> 5;            // 8 rows per pass
    const int n0 = blockIdx.x * 32;
    const int k0 = blockIdx.y * 32;
    #pragma unroll
    for (int j = 0; j < 32; j += 8)
        tile[ty + j][tx] = src[(size_t)(k0 + ty + j) * N + n0 + tx];
    __syncthreads();
    #pragma unroll
    for (int j = 0; j < 32; j += 8)
        dst[(size_t)(n0 + ty + j) * Kd + k0 + tx] = (__bf16)tile[tx][ty + j];
}

// ---------------- bf16 WMMA GEMM (both operands K-major) ----------------
// C[M,Nc] = A[M,Kc] * BT[Nc,Kc]^T + bias; OUT_MODE 0: relu->bf16, 1: f32
#define BM 128
#define BN 128
#define BK 32
#define LDSW 40   // 32 + 8 pad, keeps 16B alignment, breaks bank conflicts

template<int OUT_MODE, int Nc, int Kc>
__global__ __launch_bounds__(256)
void spline_gemm_wmma(const __bf16* __restrict__ A, const __bf16* __restrict__ BT,
                      const float* __restrict__ bias,
                      __bf16* __restrict__ outB, float* __restrict__ outF) {
    __shared__ __bf16 ldsA[2][BM * LDSW];       // [m][k], double-buffered
    __shared__ __bf16 ldsB[2][BN * LDSW];       // [n][k], double-buffered

    const int t    = threadIdx.x;
    const int lane = t & 31;
    const int wave = t >> 5;
    const int waveM = wave & 3;                 // 4 waves along M (32 rows each)
    const int waveN = wave >> 2;                // 2 waves along N (64 cols each)
    const int half = lane >> 4;
    const int l16  = lane & 15;

    const int blockM = blockIdx.x * BM;
    const int blockN = blockIdx.y * BN;

    const __bf16* Ab = A  + (size_t)blockM * Kc;
    const __bf16* Bb = BT + (size_t)blockN * Kc;

    const unsigned aBuf[2] = { (unsigned)(uintptr_t)(void*)&ldsA[0][0],
                               (unsigned)(uintptr_t)(void*)&ldsA[1][0] };
    const unsigned bBuf[2] = { (unsigned)(uintptr_t)(void*)&ldsB[0][0],
                               (unsigned)(uintptr_t)(void*)&ldsB[1][0] };

    // async copy one 128x32 bf16 tile (K-major, row stride Kc) into LDS
    auto issue_tile = [&](unsigned ldsBase, const __bf16* g, int k0) {
        #pragma unroll
        for (int it = 0; it < 2; ++it) {
            int c = t + it * 256;               // 512 chunks of 16B
            int row = c >> 2, cin = c & 3;
            unsigned ldst = ldsBase + (unsigned)((row * LDSW + cin * 8) * 2);
            unsigned long long ga = (unsigned long long)(uintptr_t)
                (g + (size_t)row * Kc + k0 + cin * 8);
            asm volatile("global_load_async_to_lds_b128 %0, %1, off"
                         :: "v"(ldst), "v"(ga) : "memory");
        }
    };

    v8f acc[2][4] = {};

    issue_tile(aBuf[0], Ab, 0);                 // prologue: tile 0 in flight
    issue_tile(bBuf[0], Bb, 0);

    constexpr int NK = Kc / BK;
    int cur = 0;
    for (int ki = 0; ki < NK; ++ki) {
        const int nxt = cur ^ 1;
        if (ki + 1 < NK) {
            // prefetch next tile (buffer last read in previous step; barrier passed)
            issue_tile(aBuf[nxt], Ab, (ki + 1) * BK);
            issue_tile(bBuf[nxt], Bb, (ki + 1) * BK);
            // async loads retire in order: <=4 outstanding => current tile landed
            asm volatile("s_wait_asynccnt 0x4" ::: "memory");
        } else {
            asm volatile("s_wait_asynccnt 0x0" ::: "memory");
        }
        __syncthreads();

        const __bf16* lA = &ldsA[cur][0];
        const __bf16* lB = &ldsB[cur][0];

        // A fragments: lane half selects K 0-7/8-15 then 16-23/24-31
        v16bf afrag[2];
        #pragma unroll
        for (int mt = 0; mt < 2; ++mt) {
            int row = waveM * 32 + mt * 16 + l16;
            v8bf lo = *(const v8bf*)&lA[row * LDSW + 8 * half];
            v8bf hi = *(const v8bf*)&lA[row * LDSW + 16 + 8 * half];
            afrag[mt] = __builtin_shufflevector(lo, hi,
                0,1,2,3,4,5,6,7,8,9,10,11,12,13,14,15);
        }
        // B fragments: lane = column, lane half selects K 0-7/8-15 then 16-23/24-31
        #pragma unroll
        for (int nt = 0; nt < 4; ++nt) {
            int col = waveN * 64 + nt * 16 + l16;
            v8bf lo = *(const v8bf*)&lB[col * LDSW + 8 * half];
            v8bf hi = *(const v8bf*)&lB[col * LDSW + 16 + 8 * half];
            v16bf bfrag = __builtin_shufflevector(lo, hi,
                0,1,2,3,4,5,6,7,8,9,10,11,12,13,14,15);
            #pragma unroll
            for (int mt = 0; mt < 2; ++mt) {
                acc[mt][nt] = __builtin_amdgcn_wmma_f32_16x16x32_bf16(
                    false, afrag[mt], false, bfrag,
                    (short)0, acc[mt][nt], false, false);
            }
        }
        __syncthreads();
        cur = nxt;
    }

    // epilogue: D layout -> lane l16 = N, VGPR r = M (r + 8*half)
    #pragma unroll
    for (int mt = 0; mt < 2; ++mt) {
        #pragma unroll
        for (int nt = 0; nt < 4; ++nt) {
            int col = blockN + waveN * 64 + nt * 16 + l16;
            float bv = bias[col];
            #pragma unroll
            for (int r = 0; r < 8; ++r) {
                int row = blockM + waveM * 32 + mt * 16 + r + 8 * half;
                float v = acc[mt][nt][r] + bv;
                if (OUT_MODE == 0) {
                    v = v > 0.f ? v : 0.f;
                    outB[(size_t)row * Nc + col] = (__bf16)v;
                } else {
                    outF[(size_t)row * Nc + col] = v;
                }
            }
        }
    }
}

// ---------------- RQ spline (one block per batch row, thread = dy) ----------------
__global__ __launch_bounds__(128)
void spline_rq_kernel(const float* __restrict__ pdf, const float* __restrict__ y,
                      float* __restrict__ out, float* __restrict__ dlogp) {
    __shared__ float coef[NOUT];                // this row's 6272 coefficients
    __shared__ float red[DY_SZ];
    const int b  = blockIdx.x;
    const int dy = threadIdx.x;
    const int t  = threadIdx.x;

    // stage pdf row into LDS with async 16B copies (coalesced stream)
    {
        const float* rowp = pdf + (size_t)b * NOUT;
        const unsigned base = (unsigned)(uintptr_t)(void*)&coef[0];
        for (int i = t; i < NOUT / 4; i += DY_SZ) {       // 1568 16B chunks
            unsigned ldst = base + (unsigned)(i * 16);
            unsigned long long ga = (unsigned long long)(uintptr_t)(rowp + i * 4);
            asm volatile("global_load_async_to_lds_b128 %0, %1, off"
                         :: "v"(ldst), "v"(ga) : "memory");
        }
        asm volatile("s_wait_asynccnt 0x0" ::: "memory");
        __syncthreads();
    }

    const float* p = &coef[dy * NCOEF];

    float w_un[K_SP], h_un[K_SP], d_un[K_SP + 1];
    #pragma unroll
    for (int k = 0; k < K_SP; ++k)     w_un[k] = p[k];
    #pragma unroll
    for (int k = 0; k < K_SP; ++k)     h_un[k] = p[K_SP + k];
    #pragma unroll
    for (int k = 0; k < K_SP + 1; ++k) d_un[k] = p[2 * K_SP + k];

    const float scl = 1.f - (float)K_SP * MIN_BIN;

    float mw = w_un[0];
    #pragma unroll
    for (int k = 1; k < K_SP; ++k) mw = fmaxf(mw, w_un[k]);
    float sw = 0.f;
    #pragma unroll
    for (int k = 0; k < K_SP; ++k) { w_un[k] = expf(w_un[k] - mw); sw += w_un[k]; }
    float rsw = 1.f / sw;

    float mh = h_un[0];
    #pragma unroll
    for (int k = 1; k < K_SP; ++k) mh = fmaxf(mh, h_un[k]);
    float sh = 0.f;
    #pragma unroll
    for (int k = 0; k < K_SP; ++k) { h_un[k] = expf(h_un[k] - mh); sh += h_un[k]; }
    float rsh = 1.f / sh;

    #pragma unroll
    for (int k = 0; k < K_SP + 1; ++k) {
        float xv = d_un[k];
        float sp = (xv > 20.f) ? xv : log1pf(expf(xv));
        d_un[k] = MIN_DERIV + sp;
    }

    float cw[K_SP + 1], ch[K_SP + 1], widths[K_SP], heights[K_SP];
    cw[0] = 0.f; ch[0] = 0.f;
    #pragma unroll
    for (int k = 0; k < K_SP; ++k) {
        widths[k]  = MIN_BIN + scl * w_un[k] * rsw;
        heights[k] = MIN_BIN + scl * h_un[k] * rsh;
        cw[k + 1] = cw[k] + widths[k];
        ch[k + 1] = ch[k] + heights[k];
    }

    const float yv = y[(size_t)b * DY_SZ + dy];

    int idx = 0;
    #pragma unroll
    for (int k = 1; k <= K_SP; ++k) idx += (yv >= cw[k]) ? 1 : 0;
    idx = idx < (K_SP - 1) ? idx : (K_SP - 1);

    const float xk  = cw[idx],      yk  = ch[idx];
    const float wk  = widths[idx],  hk  = heights[idx];
    const float dk  = d_un[idx],    dk1 = d_un[idx + 1];

    const float s   = hk / wk;
    const float tt  = (yv - xk) / wk;
    const float omt = 1.f - tt;
    const float den = s + (dk1 + dk - 2.f * s) * tt * omt;
    const float ov  = yk + hk * (s * tt * tt + dk * tt * omt) / den;
    const float der = s * s * (dk1 * tt * tt + 2.f * s * tt * omt + dk * omt * omt)
                      / (den * den);

    out[(size_t)b * DY_SZ + dy] = ov;

    red[dy] = logf(der);
    __syncthreads();
    #pragma unroll
    for (int sft = 64; sft > 0; sft >>= 1) {
        if (dy < sft) red[dy] += red[dy + sft];
        __syncthreads();
    }
    if (dy == 0) dlogp[b] = red[0];
}

// ---------------- host launch ----------------
extern "C" void kernel_launch(void* const* d_in, const int* in_sizes, int n_in,
                              void* d_out, int out_size, void* d_ws, size_t ws_size,
                              hipStream_t stream) {
    const float* x  = (const float*)d_in[0];
    const float* y  = (const float*)d_in[1];
    const float* W1 = (const float*)d_in[2];
    const float* b1 = (const float*)d_in[3];
    const float* W2 = (const float*)d_in[4];
    const float* b2 = (const float*)d_in[5];
    float* outp = (float*)d_out;

    char* ws = (char*)d_ws;
    size_t off = 0;
    __bf16* h_bf  = (__bf16*)(ws + off); off += (size_t)B_SZ * H_SZ * 2;      // 64 MiB
    __bf16* x_bf  = (__bf16*)(ws + off); off += (size_t)B_SZ * DX_SZ * 2;     // 4 MiB
    __bf16* W1T   = (__bf16*)(ws + off); off += (size_t)DX_SZ * H_SZ * 2;     // 0.5 MiB  [H][DX]
    __bf16* W2T   = (__bf16*)(ws + off); off += (size_t)H_SZ * NOUT * 2;      // 24.5 MiB [NOUT][H]
    float*  pdf   = (float*)(ws + off);                                       // 392 MiB

    // x -> bf16; weights -> bf16 transposed (N-major, K contiguous)
    spline_cvt_bf16<<<(B_SZ * DX_SZ + 255) / 256, 256, 0, stream>>>(x, x_bf, B_SZ * DX_SZ);
    spline_cvtT_bf16<<<dim3(H_SZ / 32, DX_SZ / 32), 256, 0, stream>>>(W1, W1T, DX_SZ, H_SZ);
    spline_cvtT_bf16<<<dim3(NOUT / 32, H_SZ / 32), 256, 0, stream>>>(W2, W2T, H_SZ, NOUT);

    // h = relu(x @ W1 + b1) -> bf16
    dim3 g1(B_SZ / BM, H_SZ / BN);
    spline_gemm_wmma<0, H_SZ, DX_SZ><<<g1, 256, 0, stream>>>(x_bf, W1T, b1, h_bf, nullptr);
    // pdf = h @ W2 + b2 -> f32
    dim3 g2(B_SZ / BM, NOUT / BN);   // 6272 = 49 * 128
    spline_gemm_wmma<1, NOUT, H_SZ><<<g2, 256, 0, stream>>>(h_bf, W2T, b2, nullptr, pdf);
    // spline transform + dlogp reduction
    float* dlogp = outp + (size_t)B_SZ * DY_SZ;
    spline_rq_kernel<<<B_SZ, DY_SZ, 0, stream>>>(pdf, y, outp, dlogp);
}
